// TransformerEncoderLayer_38938173505640
// MI455X (gfx1250) — matrix operand order repaired and checked by reading, not compiled, not run on hardware
//
#include <hip/hip_runtime.h>
#include <hip/hip_bf16.h>
#include <cstdint>
#include <cstddef>

// ---------------------------------------------------------------------------
// MI455X / gfx1250 transformer encoder layer.
// Compute-bound (~170 GFLOP vs ~100MB traffic) -> everything matrix-shaped
// runs on v_wmma_f32_16x16x32_f16 (wave32 WMMA, f32 accumulate).
// Attention is single-pass flash-style so S x S scores never touch HBM.
// GEMM tile fills use gfx1250 async global->LDS loads (ASYNCcnt path).
// Softmax row-max uses DPP16 butterflies (no LDS); row-sum uses a WMMA
// against a ones fragment (C layout == per-row stats layout).
// ---------------------------------------------------------------------------

typedef __attribute__((ext_vector_type(16))) _Float16 v16h;
typedef __attribute__((ext_vector_type(8)))  _Float16 v8h;
typedef __attribute__((ext_vector_type(8)))  float    v8f;

constexpr int Dm   = 768;    // hidden
constexpr int Hh   = 12;     // heads
constexpr int HDim = 64;     // head dim
constexpr int Ff   = 3072;   // ffn
constexpr int Bb   = 4;      // batch
constexpr int Ss   = 2048;   // seq
constexpr int Mrows = Bb * Ss;   // 8192 token rows

static __device__ __forceinline__ v16h concat8(v8h lo, v8h hi) {
  return __builtin_shufflevector(lo, hi, 0,1,2,3,4,5,6,7,8,9,10,11,12,13,14,15);
}

static __device__ __forceinline__ v8f wmma_f16(v16h a, v16h b, v8f c) {
  // D = A(16x32 f16) * B(32x16 f16) + C(16x16 f32)
  return __builtin_amdgcn_wmma_f32_16x16x32_f16(false, a, false, b, (short)0, c,
                                                false, false);
}

// gfx1250 async global->LDS copy, 16B per lane; tracked by ASYNCcnt.
// LDS destination address = low 32 bits of the generic pointer (ISA 10.2).
static __device__ __forceinline__ void async_load_b128(uint32_t lds_off,
                                                       uint64_t gaddr) {
  asm volatile("global_load_async_to_lds_b128 %0, %1, off"
               :: "v"(lds_off), "v"(gaddr)
               : "memory");
}
static __device__ __forceinline__ void wait_async0() {
  asm volatile("s_wait_asynccnt 0" ::: "memory");
}
static __device__ __forceinline__ uint32_t lds_off32(const void* p) {
  return (uint32_t)(uintptr_t)p;   // addr[31:0] == LDS byte offset
}

// Allreduce-max across each 16-lane half via DPP16 (no LDS traffic).
// Steps: quad swap1, quad swap2, half-row mirror, row mirror.
static __device__ __forceinline__ float dpp_max16(float x) {
  int t;
  t = __builtin_amdgcn_update_dpp(0, __float_as_int(x), 0xB1,  0xF, 0xF, true);
  x = fmaxf(x, __int_as_float(t));   // quad_perm(1,0,3,2)
  t = __builtin_amdgcn_update_dpp(0, __float_as_int(x), 0x4E,  0xF, 0xF, true);
  x = fmaxf(x, __int_as_float(t));   // quad_perm(2,3,0,1)
  t = __builtin_amdgcn_update_dpp(0, __float_as_int(x), 0x141, 0xF, 0xF, true);
  x = fmaxf(x, __int_as_float(t));   // row_half_mirror
  t = __builtin_amdgcn_update_dpp(0, __float_as_int(x), 0x140, 0xF, 0xF, true);
  x = fmaxf(x, __int_as_float(t));   // row_mirror
  return x;
}

// ---------------------------------------------------------------------------
// Weight prep: f32 -> f16, pre-transposed to [N][K] so WMMA B-fragments are
// two contiguous 8-half k runs per lane.
// ---------------------------------------------------------------------------
__global__ void cvt_wqkv_kernel(const float* __restrict__ Wq,
                                const float* __restrict__ Wk,
                                const float* __restrict__ Wv,
                                _Float16* __restrict__ Wt) {
  const int idx = blockIdx.x * blockDim.x + threadIdx.x;
  constexpr int per = Hh * Dm * HDim;
  if (idx >= 3 * per) return;
  const int p = idx / per, r = idx - p * per;
  const int hh = r / (Dm * HDim);
  const int rem = r - hh * (Dm * HDim);
  const int d = rem / HDim, e = rem - d * HDim;
  const float* W = (p == 0) ? Wq : (p == 1) ? Wk : Wv;
  const int n = p * Dm + hh * HDim + e;        // column in combined QKV
  Wt[(size_t)n * Dm + d] = (_Float16)W[r];     // [N=2304][K=768]
}

__global__ void transpose_f32_to_f16(const float* __restrict__ src,  // [K][N]
                                     _Float16* __restrict__ dst,     // [N][K]
                                     int K, int N) {
  const int idx = blockIdx.x * blockDim.x + threadIdx.x;
  if (idx >= K * N) return;
  const int kk = idx / N, n = idx - kk * N;
  dst[(size_t)n * K + kk] = (_Float16)src[idx];
}

// ---------------------------------------------------------------------------
// LayerNorm over D=768, one block per token row, f32 in -> f16 out.
// ---------------------------------------------------------------------------
__global__ __launch_bounds__(256) void ln_kernel(const float* __restrict__ x,
                                                 const float* __restrict__ gamma,
                                                 const float* __restrict__ beta,
                                                 _Float16* __restrict__ y) {
  const int row = blockIdx.x, tid = threadIdx.x;
  const float* xr = x + (size_t)row * Dm;
  const float v0 = xr[tid], v1 = xr[tid + 256], v2 = xr[tid + 512];
  float s  = v0 + v1 + v2;
  float s2 = v0 * v0 + v1 * v1 + v2 * v2;
#pragma unroll
  for (int off = 1; off < 32; off <<= 1) {
    s  += __shfl_xor(s,  off, 32);
    s2 += __shfl_xor(s2, off, 32);
  }
  __shared__ float red1[8], red2[8];
  const int wave = tid >> 5, lane = tid & 31;
  if (lane == 0) { red1[wave] = s; red2[wave] = s2; }
  __syncthreads();
  if (tid < 32) {
    float a1 = (lane < 8) ? red1[lane] : 0.f;
    float a2 = (lane < 8) ? red2[lane] : 0.f;
#pragma unroll
    for (int off = 1; off < 8; off <<= 1) {
      a1 += __shfl_xor(a1, off, 32);
      a2 += __shfl_xor(a2, off, 32);
    }
    if (lane == 0) { red1[0] = a1; red2[0] = a2; }
  }
  __syncthreads();
  const float mu   = red1[0] * (1.0f / Dm);
  const float var  = red2[0] * (1.0f / Dm) - mu * mu;
  const float rstd = rsqrtf(var + 1e-5f);
  _Float16* yr = y + (size_t)row * Dm;
  yr[tid]       = (_Float16)((v0 - mu) * rstd * gamma[tid]       + beta[tid]);
  yr[tid + 256] = (_Float16)((v1 - mu) * rstd * gamma[tid + 256] + beta[tid + 256]);
  yr[tid + 512] = (_Float16)((v2 - mu) * rstd * gamma[tid + 512] + beta[tid + 512]);
}

// ---------------------------------------------------------------------------
// Tiled WMMA GEMM: C[M,N] = A[M,K](f16) * Bt[N,K]^T(f16), fused epilogues.
// Block = 256 threads (8 waves), tile 64x64, K-step 32.
// Tile fill: one async b128 global->LDS transfer per thread per operand.
// Wave w: row-tile (w&3), col-tiles {2*(w>>2), 2*(w>>2)+1}.
// ---------------------------------------------------------------------------
constexpr int EPI_QKV      = 0;  // bias per proj + scatter to q/k/vt (f16)
constexpr int EPI_BIAS_F32 = 1;  // +bias, store f32
constexpr int EPI_GELU_F16 = 2;  // +bias, exact-erf GELU, store f16

template <int EPI>
__global__ __launch_bounds__(256) void gemm_kernel(
    const _Float16* __restrict__ A, const _Float16* __restrict__ Bt,
    const int N, const int K,
    const float* __restrict__ bias0, const float* __restrict__ bias1,
    const float* __restrict__ bias2,
    float* __restrict__ out_f, _Float16* __restrict__ out_h0,
    _Float16* __restrict__ out_h1, _Float16* __restrict__ out_h2) {
  __shared__ __align__(16) _Float16 As[64 * 40];  // [m][k], pad to 40 halves
  __shared__ __align__(16) _Float16 Bs[64 * 40];  // [n][k]
  const int tid   = threadIdx.x;
  const int wave  = tid >> 5, lane = tid & 31;
  const int lhalf = lane >> 4, lcol = lane & 15;
  const int n0 = blockIdx.x * 64, m0 = blockIdx.y * 64;
  const int ldrow = tid >> 2, ldcol = (tid & 3) * 8;
  const int rtile = wave & 3, cbase = (wave >> 2) * 2;

  const uint32_t ldsA = lds_off32(&As[ldrow * 40 + ldcol]);
  const uint32_t ldsB = lds_off32(&Bs[ldrow * 40 + ldcol]);
  const _Float16* gA = A  + (size_t)(m0 + ldrow) * K + ldcol;
  const _Float16* gB = Bt + (size_t)(n0 + ldrow) * K + ldcol;

  v8f zacc = {};
  v8f accs[2] = {zacc, zacc};

  for (int kt = 0; kt < K; kt += 32) {
    // async tile fill: 16B/lane direct to LDS (ASYNCcnt tracked)
    async_load_b128(ldsA, (uint64_t)(uintptr_t)(gA + kt));
    async_load_b128(ldsB, (uint64_t)(uintptr_t)(gB + kt));
    // prefetch next K-tile toward WGP$ while this tile computes
    if (kt + 32 < K) {
      __builtin_prefetch(gA + kt + 32, 0, 0);
      __builtin_prefetch(gB + kt + 32, 0, 0);
    }
    wait_async0();
    __syncthreads();

    // A fragment: lane holds row m=lane&15; halves 0..7 -> k = 8*lhalf..,
    // halves 8..15 -> k = 16+8*lhalf.. (per gfx1250 16-bit A layout)
    const int mm = rtile * 16 + lcol;
    const v16h af = concat8(*(const v8h*)&As[mm * 40 + lhalf * 8],
                            *(const v8h*)&As[mm * 40 + 16 + lhalf * 8]);
#pragma unroll
    for (int c = 0; c < 2; ++c) {
      // B fragment: lane holds col n=lane&15, k = 8*lhalf.. and 16+8*lhalf.. (matches A)
      const int nn = (cbase + c) * 16 + lcol;
      const v16h bf = concat8(*(const v8h*)&Bs[nn * 40 + lhalf * 8],
                              *(const v8h*)&Bs[nn * 40 + 16 + lhalf * 8]);
      accs[c] = wmma_f16(af, bf, accs[c]);
    }
    __syncthreads();
  }

  // C layout: vgpr r -> row (r + 8*lhalf), lane col n = lane&15
#pragma unroll
  for (int c = 0; c < 2; ++c) {
    const int n = n0 + (cbase + c) * 16 + lcol;
#pragma unroll
    for (int r = 0; r < 8; ++r) {
      const int m = m0 + rtile * 16 + r + 8 * lhalf;
      const float val = accs[c][r];
      if constexpr (EPI == EPI_QKV) {
        const int p = n / Dm, nn2 = n - p * Dm;
        const float bsv = (p == 0 ? bias0 : (p == 1 ? bias1 : bias2))[nn2];
        const _Float16 hv = (_Float16)(val + bsv);
        const int hh = nn2 >> 6, e = nn2 & 63;
        const int bbi = m >> 11, srow = m & (Ss - 1);
        const size_t bh = (size_t)bbi * Hh + hh;
        if (p == 0)      out_h0[(bh * Ss + srow) * HDim + e] = hv;  // Q [bh][s][e]
        else if (p == 1) out_h1[(bh * Ss + srow) * HDim + e] = hv;  // K [bh][s][e]
        else             out_h2[(bh * HDim + e) * Ss + srow] = hv;  // V^T [bh][e][s]
      } else if constexpr (EPI == EPI_BIAS_F32) {
        out_f[(size_t)m * N + n] = val + bias0[n];
      } else {  // EPI_GELU_F16, exact erf GELU
        const float xg = val + bias0[n];
        out_h0[(size_t)m * N + n] =
            (_Float16)(0.5f * xg * (1.0f + erff(xg * 0.70710678118654752f)));
      }
    }
  }
}

// ---------------------------------------------------------------------------
// Flash attention: block = 8 waves, each wave owns 16 queries of one (b,h),
// streams keys in 32-wide tiles with online softmax. 9 WMMAs / tile / wave
// (4 score + 1 rowsum-of-P + 4 P*V). Row max via DPP16 butterflies (VALU
// only, co-executes with XDL WMMAs); LDS used only for the P re-layout.
// q,k: [bh][s][64] f16 ; vt: [bh][64][s] f16 ; o: [b][s][D] f16 (heads concat)
// ---------------------------------------------------------------------------
__global__ __launch_bounds__(256) void attn_kernel(const _Float16* __restrict__ q,
                                                   const _Float16* __restrict__ k,
                                                   const _Float16* __restrict__ vt,
                                                   _Float16* __restrict__ o) {
  __shared__ __align__(16) _Float16 Ps[8 * 16 * 40];  // per-wave P bounce buffer
  const int tid   = threadIdx.x;
  const int wave  = tid >> 5, lane = tid & 31;
  const int lhalf = lane >> 4, lcol = lane & 15;
  constexpr int nqblk = Ss / 128;  // 16
  const int bh   = blockIdx.x / nqblk;
  const int qblk = blockIdx.x - bh * nqblk;
  const int m0   = qblk * 128 + wave * 16;  // query row within (b,h)
  _Float16* Pw = &Ps[wave * 16 * 40];

  // Q fragments (A-matrix, 16x64 split into two 16x32 K-chunks), kept
  // resident; 1/sqrt(HD)=0.125 folded in here (exact in f16).
  const size_t qbase = ((size_t)bh * Ss + (m0 + lcol)) * HDim;
  v16h qa0 = concat8(*(const v8h*)(q + qbase + lhalf * 8),
                     *(const v8h*)(q + qbase + 16 + lhalf * 8));
  v16h qa1 = concat8(*(const v8h*)(q + qbase + 32 + lhalf * 8),
                     *(const v8h*)(q + qbase + 48 + lhalf * 8));
  v16h sc, ones;
#pragma unroll
  for (int i = 0; i < 16; ++i) { sc[i] = (_Float16)0.125f; ones[i] = (_Float16)1.0f; }
  qa0 *= sc;
  qa1 *= sc;

  float mi[8], li[8], alr[8];
  v8f zacc = {};
  v8f oacc[4] = {zacc, zacc, zacc, zacc};
#pragma unroll
  for (int r = 0; r < 8; ++r) { mi[r] = -3.0e38f; li[r] = 0.f; }

  const size_t kb_base = (size_t)bh * Ss * HDim;
  const size_t vb_base = (size_t)bh * HDim * Ss;

  for (int t = 0; t < Ss / 32; ++t) {
    // scores = (Q/8) * K^T for 32 keys: two 16x16 C tiles, 2 K-steps each
    v8f c0 = {}, c1 = {};
    {
      const size_t kr0 = kb_base + (size_t)(t * 32 + lcol) * HDim + lhalf * 8;
      const size_t kr1 = kb_base + (size_t)(t * 32 + 16 + lcol) * HDim + lhalf * 8;
      const v16h kb00 = concat8(*(const v8h*)(k + kr0),      *(const v8h*)(k + kr0 + 16));
      const v16h kb01 = concat8(*(const v8h*)(k + kr0 + 32), *(const v8h*)(k + kr0 + 48));
      const v16h kb10 = concat8(*(const v8h*)(k + kr1),      *(const v8h*)(k + kr1 + 16));
      const v16h kb11 = concat8(*(const v8h*)(k + kr1 + 32), *(const v8h*)(k + kr1 + 48));
      c0 = wmma_f16(qa0, kb00, c0);
      c0 = wmma_f16(qa1, kb01, c0);
      c1 = wmma_f16(qa0, kb10, c1);
      c1 = wmma_f16(qa1, kb11, c1);
    }

    // online softmax over this 32-key tile (rows split across 16-lane halves)
#pragma unroll
    for (int r = 0; r < 8; ++r) {
      const float v0 = c0[r];
      const float v1 = c1[r];
      const float tm = dpp_max16(fmaxf(v0, v1));   // VALU-only 16-lane allreduce
      const float mnew = fmaxf(mi[r], tm);
      const float al = __expf(mi[r] - mnew);
      const float p0 = __expf(v0 - mnew);
      const float p1 = __expf(v1 - mnew);
      mi[r] = mnew;
      alr[r] = al;
#pragma unroll
      for (int e = 0; e < 4; ++e) oacc[e][r] *= al;
      // stash P in C-layout -> LDS (row m = r+8*lhalf, key col = lane&15)
      Pw[(r + 8 * lhalf) * 40 + lcol]      = (_Float16)p0;
      Pw[(r + 8 * lhalf) * 40 + 16 + lcol] = (_Float16)p1;
    }
    asm volatile("s_wait_dscnt 0" ::: "memory");  // wave-private LDS RAW fence

    // regather P as an A fragment (16x32)
    const v16h pa = concat8(*(const v8h*)&Pw[lcol * 40 + lhalf * 8],
                            *(const v8h*)&Pw[lcol * 40 + 16 + lhalf * 8]);

    // row sums of P via WMMA against ones: C layout == stats layout
    v8f zs = {};
    const v8f ts = wmma_f16(pa, ones, zs);
#pragma unroll
    for (int r = 0; r < 8; ++r) li[r] = li[r] * alr[r] + ts[r];

    // O += P * V : V^T rows are contiguous per (e, key-chunk)
#pragma unroll
    for (int e = 0; e < 4; ++e) {
      const size_t vr = vb_base + (size_t)(e * 16 + lcol) * Ss + t * 32 + lhalf * 8;
      const v16h vb = concat8(*(const v8h*)(vt + vr), *(const v8h*)(vt + vr + 16));
      oacc[e] = wmma_f16(pa, vb, oacc[e]);
    }
  }

  // normalize and scatter into concat-head layout o[b][s][h*64+e]
  const int bbi = bh / Hh, hh = bh - bbi * Hh;
#pragma unroll
  for (int r = 0; r < 8; ++r) {
    const float inv = 1.0f / li[r];
    const int srow = m0 + r + 8 * lhalf;
    const size_t obase = ((size_t)bbi * Ss + srow) * Dm + hh * HDim;
#pragma unroll
    for (int e = 0; e < 4; ++e)
      o[obase + e * 16 + lcol] = (_Float16)(oacc[e][r] * inv);
  }
}

// ---------------------------------------------------------------------------
// Workspace layout (bytes). Light reuse keeps total ~115 MB.
// ---------------------------------------------------------------------------
constexpr size_t SZ_ACT   = (size_t)Mrows * Dm * 2;      // 12.58 MB f16 activation
constexpr size_t OFF_WQKVT = 0;                                     // 3*768*768 f16
constexpr size_t OFF_WOT   = OFF_WQKVT + (size_t)3 * Dm * Dm * 2;   // 768x768 f16
constexpr size_t OFF_W1T   = OFF_WOT   + (size_t)Dm * Dm * 2;       // 3072x768 f16
constexpr size_t OFF_W2T   = OFF_W1T   + (size_t)Dm * Ff * 2;       // 768x3072 f16
constexpr size_t OFF_H1    = OFF_W2T   + (size_t)Ff * Dm * 2;       // h1 (also o_buf)
constexpr size_t OFF_Q     = OFF_H1 + SZ_ACT;
constexpr size_t OFF_K     = OFF_Q  + SZ_ACT;
constexpr size_t OFF_VT    = OFF_K  + SZ_ACT;                       // vt (later h2)
constexpr size_t OFF_F     = OFF_VT + SZ_ACT;                       // f16 [8192][3072]
constexpr size_t OFF_A32   = OFF_Q;   // f32 'a' overlays q+k (25.2MB <= 2*12.6MB)

extern "C" void kernel_launch(void* const* d_in, const int* in_sizes, int n_in,
                              void* d_out, int out_size, void* d_ws, size_t ws_size,
                              hipStream_t stream) {
  (void)in_sizes; (void)n_in; (void)out_size; (void)ws_size;
  const float* x   = (const float*)d_in[0];
  const float* Wq  = (const float*)d_in[1];
  const float* bq  = (const float*)d_in[2];
  const float* Wk  = (const float*)d_in[3];
  const float* bk  = (const float*)d_in[4];
  const float* Wv  = (const float*)d_in[5];
  const float* bv  = (const float*)d_in[6];
  const float* Wo  = (const float*)d_in[7];
  const float* bo  = (const float*)d_in[8];
  const float* W1  = (const float*)d_in[9];
  const float* b1  = (const float*)d_in[10];
  const float* W2  = (const float*)d_in[11];
  const float* b2  = (const float*)d_in[12];
  const float* g1  = (const float*)d_in[13];
  const float* be1 = (const float*)d_in[14];
  const float* g2  = (const float*)d_in[15];
  const float* be2 = (const float*)d_in[16];
  float* out = (float*)d_out;

  char* ws = (char*)d_ws;
  _Float16* wqkvt = (_Float16*)(ws + OFF_WQKVT);
  _Float16* wot   = (_Float16*)(ws + OFF_WOT);
  _Float16* w1t   = (_Float16*)(ws + OFF_W1T);
  _Float16* w2t   = (_Float16*)(ws + OFF_W2T);
  _Float16* h1    = (_Float16*)(ws + OFF_H1);
  _Float16* qb    = (_Float16*)(ws + OFF_Q);
  _Float16* kb    = (_Float16*)(ws + OFF_K);
  _Float16* vtb   = (_Float16*)(ws + OFF_VT);
  _Float16* obuf  = (_Float16*)(ws + OFF_H1);   // reuse h1 region
  float*    a32   = (float*)   (ws + OFF_A32);  // reuse q+k region
  _Float16* h2    = (_Float16*)(ws + OFF_VT);   // reuse vt region
  _Float16* fbuf  = (_Float16*)(ws + OFF_F);

  // --- weight prep (f32 -> f16 transposed) ---
  cvt_wqkv_kernel<<<(3 * Hh * Dm * HDim + 255) / 256, 256, 0, stream>>>(Wq, Wk, Wv, wqkvt);
  transpose_f32_to_f16<<<(Dm * Dm + 255) / 256, 256, 0, stream>>>(Wo, wot, Dm, Dm);
  transpose_f32_to_f16<<<(Dm * Ff + 255) / 256, 256, 0, stream>>>(W1, w1t, Dm, Ff);
  transpose_f32_to_f16<<<(Ff * Dm + 255) / 256, 256, 0, stream>>>(W2, w2t, Ff, Dm);

  // --- LN1 ---
  ln_kernel<<<Mrows, 256, 0, stream>>>(x, g1, be1, h1);

  // --- fused QKV projection (N = 3*768 = 2304) ---
  gemm_kernel<EPI_QKV><<<dim3(2304 / 64, Mrows / 64), 256, 0, stream>>>(
      h1, wqkvt, 2304, Dm, bq, bk, bv, nullptr, qb, kb, vtb);

  // --- flash attention ---
  attn_kernel<<<Bb * Hh * (Ss / 128), 256, 0, stream>>>(qb, kb, vtb, obuf);

  // --- output projection -> a (f32) ---
  gemm_kernel<EPI_BIAS_F32><<<dim3(Dm / 64, Mrows / 64), 256, 0, stream>>>(
      obuf, wot, Dm, Dm, bo, nullptr, nullptr, a32, nullptr, nullptr, nullptr);

  // --- LN2 ---
  ln_kernel<<<Mrows, 256, 0, stream>>>(a32, g2, be2, h2);

  // --- FFN1 + exact GELU -> f (f16) ---
  gemm_kernel<EPI_GELU_F16><<<dim3(Ff / 64, Mrows / 64), 256, 0, stream>>>(
      h2, w1t, Ff, Dm, b1, nullptr, nullptr, nullptr, fbuf, nullptr, nullptr);

  // --- FFN2 -> output (f32) ---
  gemm_kernel<EPI_BIAS_F32><<<dim3(Dm / 64, Mrows / 64), 256, 0, stream>>>(
      fbuf, w2t, Dm, Ff, b2, nullptr, nullptr, out, nullptr, nullptr, nullptr);
}
